// Capsule_566935683098
// MI455X (gfx1250) — matrix operand order repaired and checked by reading, not compiled, not run on hardware
//
#include <hip/hip_runtime.h>
#include <hip/hip_bf16.h>
#include <stdint.h>

// ---------------------------------------------------------------------------
// Capsule routing for MI455X (gfx1250, wave32).
//   u_hat GEMM: M=32768 (B*S), K=512, N=2048 (32 caps * 64 dim)
//   -> v_wmma_f32_16x16x32_bf16, f32 accumulate, transpose folded into store,
//      double-buffered LDS fed by GLOBAL_LOAD_ASYNC_TO_LDS_B128 (ASYNCcnt).
//   Routing: bandwidth-bound streaming kernels over u_hat (fp32).
// ---------------------------------------------------------------------------

typedef __attribute__((ext_vector_type(16))) __bf16 v16bf;
typedef __attribute__((ext_vector_type(8)))  __bf16 v8bf;
typedef __attribute__((ext_vector_type(8)))  float  v8f;
typedef int v4i __attribute__((vector_size(16)));   // matches builtin param type

#define NCAP 32
#define DCAP 64
#define BATCH 64
#define SEQ 512
#define DIN 512
#define NCOL (NCAP * DCAP)   // 2048
#define MROW (BATCH * SEQ)   // 32768

// ------------------------- CDNA5 async-copy plumbing -----------------------
#if __has_builtin(__builtin_amdgcn_global_load_async_to_lds_b128)
#define HAVE_ASYNC_LDS 1
__device__ __forceinline__ void async_copy16(const void* g, void* l) {
    __builtin_amdgcn_global_load_async_to_lds_b128(
        (__attribute__((address_space(1))) v4i*)const_cast<void*>(g),
        (__attribute__((address_space(3))) v4i*)l, 0, 0);
}
#else
#define HAVE_ASYNC_LDS 0
__device__ __forceinline__ void async_copy16(const void* g, void* l) {
    *(uint4*)l = *(const uint4*)g;   // synchronous fallback
}
#endif

#if __has_builtin(__builtin_amdgcn_s_wait_asynccnt)
#define WAIT_ASYNCCNT(n) __builtin_amdgcn_s_wait_asynccnt(n)
#elif HAVE_ASYNC_LDS
#define WAIT_ASYNCCNT(n) asm volatile("s_wait_asynccnt %0" :: "i"(n) : "memory")
#else
#define WAIT_ASYNCCNT(n) do {} while (0)
#endif

__device__ __forceinline__ unsigned short f2bf(float f) {
    unsigned int u = __float_as_uint(f);
    unsigned int r = (u + 0x7FFFu + ((u >> 16) & 1u)) >> 16;  // RNE
    return (unsigned short)r;
}

// --------------------------- fp32 -> bf16 convert ---------------------------
__global__ void cvt_bf16x4(const float* __restrict__ in,
                           unsigned short* __restrict__ out, int n4) {
    int idx = blockIdx.x * blockDim.x + threadIdx.x;
    int stride = gridDim.x * blockDim.x;
    for (int i = idx; i < n4; i += stride) {
        float4 v = *(const float4*)(in + (size_t)i * 4);
        unsigned int p0 = f2bf(v.x) | ((unsigned int)f2bf(v.y) << 16);
        unsigned int p1 = f2bf(v.z) | ((unsigned int)f2bf(v.w) << 16);
        ((uint2*)out)[i] = make_uint2(p0, p1);
    }
}

// ------------------- fp32 W [K][N] -> bf16 W^T [N][K] ----------------------
// One-time 4 MB transform; lets the GEMM load B tiles as contiguous K runs.
__global__ void cvt_w_transpose(const float* __restrict__ W,
                                unsigned short* __restrict__ Wt) {
    int idx = blockIdx.x * blockDim.x + threadIdx.x;   // over NCOL*DIN
    if (idx >= NCOL * DIN) return;
    int n = idx >> 9;            // / DIN
    int k = idx & (DIN - 1);
    Wt[idx] = f2bf(W[(size_t)k * NCOL + n]);
}

// ------------------------------- WMMA GEMM ---------------------------------
// Block tile 128x128, K-step 32, double-buffered async LDS pipeline.
// 8 waves (256 thr); wave grid 4(M) x 2(N); wave owns 32x64 = 2x4 WMMA tiles.
// Output written directly in (B, NCAP, S, DCAP) layout.
#define BM 128
#define BN 128
#define BK 32

__global__ __launch_bounds__(256) void gemm_uhat(
        const unsigned short* __restrict__ A,   // bf16 bits, [MROW][DIN]
        const unsigned short* __restrict__ Wt,  // bf16 bits, [NCOL][DIN]
        float* __restrict__ uhat) {             // [B][NCAP][S][DCAP]
    __shared__ __bf16 As[2][BM][BK];   // 2 x 8 KB
    __shared__ __bf16 Bs[2][BN][BK];   // 2 x 8 KB, already K-major (from W^T)

    const int tid  = threadIdx.x;
    const int lane = tid & 31;
    const int wave = tid >> 5;
    const int rowBase = blockIdx.x * BM;
    const int colBase = blockIdx.y * BN;
    const int wm = wave >> 1;       // 0..3
    const int wn = wave & 1;        // 0..1

    // A-fragment lane addressing (ISA 7.12.2, 16-bit A 16x32):
    //   lanes 0-15: K in [0,8)+[16,24); lanes 16-31: K in [8,16)+[24,32)
    const int am  = lane & 15;
    const int akb = (lane >> 4) * 8;     // 0 or 8
    // B-fragment (K-major per lane): lanes 0-15: K in [0,8)+[16,24); lanes 16-31: K in [8,16)+[24,32)
    const int bnn = lane & 15;
    const int bkb = (lane >> 4) * 8;     // 0 or 8

    // per-stage copy: each thread moves 32 B of A and 32 B of B (4 async b128)
    const int cr  = tid >> 1;            // 0..127 (row of A tile / row of B tile)
    const int ckk = (tid & 1) * 16;      // 0 or 16 (K offset)
    const unsigned short* aSrc = A  + (size_t)(rowBase + cr) * DIN + ckk;
    const unsigned short* bSrc = Wt + (size_t)(colBase + cr) * DIN + ckk;

    v8f acc[2][4] = {};

    auto issue_stage = [&](int k0, int buf) {
        async_copy16(aSrc + k0,     &As[buf][cr][ckk]);
        async_copy16(aSrc + k0 + 8, &As[buf][cr][ckk + 8]);
        async_copy16(bSrc + k0,     &Bs[buf][cr][ckk]);
        async_copy16(bSrc + k0 + 8, &Bs[buf][cr][ckk + 8]);
    };

    const int NK = DIN / BK;   // 16
    issue_stage(0, 0);
    for (int ik = 0; ik < NK; ++ik) {
        const int cur = ik & 1;
        if (ik + 1 < NK) issue_stage((ik + 1) * BK, 1 - cur);
        // in-order completion: <=4 outstanding leaves only next stage in flight
        if (ik + 1 < NK) { WAIT_ASYNCCNT(4); } else { WAIT_ASYNCCNT(0); }
        __syncthreads();

        v16bf afrag[2], bfrag[4];
        #pragma unroll
        for (int i = 0; i < 2; ++i) {
            int m = wm * 32 + i * 16 + am;
            v8bf lo = *(const v8bf*)&As[cur][m][akb];
            v8bf hi = *(const v8bf*)&As[cur][m][akb + 16];
            afrag[i] = __builtin_shufflevector(lo, hi,
                0,1,2,3,4,5,6,7,8,9,10,11,12,13,14,15);
        }
        #pragma unroll
        for (int j = 0; j < 4; ++j) {
            int n = wn * 64 + j * 16 + bnn;
            v8bf lo = *(const v8bf*)&Bs[cur][n][bkb];
            v8bf hi = *(const v8bf*)&Bs[cur][n][bkb + 16];
            bfrag[j] = __builtin_shufflevector(lo, hi,
                0,1,2,3,4,5,6,7,8,9,10,11,12,13,14,15);
        }
        #pragma unroll
        for (int i = 0; i < 2; ++i)
            #pragma unroll
            for (int j = 0; j < 4; ++j)
                acc[i][j] = __builtin_amdgcn_wmma_f32_16x16x32_bf16(
                    false, afrag[i], false, bfrag[j],
                    (short)0, acc[i][j], false, false);
        __syncthreads();   // all waves done reading buf[cur] before its reuse
    }

    // Store: C/D layout (f32 16x16): col = lane&15, row = v + 8*(lane>>4).
    // Fold transpose: row R -> (b, s); col C -> (ncap, d).
    const int b = rowBase >> 9;   // constant per block (BM=128, SEQ=512)
    #pragma unroll
    for (int i = 0; i < 2; ++i) {
        int tm = wm * 32 + i * 16;
        #pragma unroll
        for (int j = 0; j < 4; ++j) {
            int tn   = colBase + wn * 64 + j * 16;
            int ncap = tn >> 6;
            int dd   = (tn & 63) + (lane & 15);
            float* dst = uhat + ((size_t)b * NCAP + ncap) * SEQ * DCAP;
            #pragma unroll
            for (int v = 0; v < 8; ++v) {
                int s = (rowBase + tm + v + ((lane >> 4) << 3)) & (SEQ - 1);
                dst[(size_t)s * DCAP + dd] = acc[i][j][v];
            }
        }
    }
}

// --------------------- outputs = squash(sum_i c * u_hat) -------------------
// One block per (b, n). c == nullptr means uniform coupling (iteration 0;
// squash L2-normalizes so the 1/32 factor is irrelevant).
__global__ __launch_bounds__(256) void k_out(const float* __restrict__ uhat,
                                             const float* __restrict__ c,
                                             float* __restrict__ outv) {
    const int bn = blockIdx.x;          // b*32 + n
    const int t  = threadIdx.x;
    const int d  = t & 63;
    const int ic = t >> 6;              // 0..3
    const float* up = uhat + (size_t)bn * SEQ * DCAP;

    float acc = 0.f;
    if (c) {
        const float* cp = c + (size_t)bn * SEQ;
        for (int i = ic; i < SEQ; i += 4) acc += cp[i] * up[(size_t)i * DCAP + d];
    } else {
        for (int i = ic; i < SEQ; i += 4) acc += up[(size_t)i * DCAP + d];
    }

    __shared__ float part[4][DCAP];
    __shared__ float red[DCAP];
    __shared__ float scale;
    part[ic][d] = acc;
    __syncthreads();
    if (t < DCAP) red[t] = part[0][t] + part[1][t] + part[2][t] + part[3][t];
    __syncthreads();
    if (t == 0) {
        float s = 0.f;
        #pragma unroll
        for (int q = 0; q < DCAP; ++q) s += red[q] * red[q];
        scale = rsqrtf(s + 1e-7f);      // squash == x / sqrt(ss + eps)
    }
    __syncthreads();
    if (t < DCAP) outv[(size_t)bn * DCAP + t] = red[t] * scale;
}

// ------------------- b[b,n,i] = <outputs[b,n,:], u_hat[b,n,i,:]> -----------
__global__ __launch_bounds__(256) void k_bupd(const float* __restrict__ uhat,
                                              const float* __restrict__ outv,
                                              float* __restrict__ barr) {
    const int bn = blockIdx.x;
    const int t  = threadIdx.x;
    __shared__ float ov[DCAP];
    if (t < DCAP) ov[t] = outv[(size_t)bn * DCAP + t];
    __syncthreads();
    const float* up = uhat + (size_t)bn * SEQ * DCAP;
    for (int i = t; i < SEQ; i += 256) {
        const float4* row = (const float4*)(up + (size_t)i * DCAP);
        float s = 0.f;
        #pragma unroll
        for (int q = 0; q < 16; ++q) {
            float4 v = row[q];
            s += v.x * ov[q*4+0] + v.y * ov[q*4+1]
               + v.z * ov[q*4+2] + v.w * ov[q*4+3];
        }
        barr[(size_t)bn * SEQ + i] = s;
    }
}

// ---------------- softmax over capsule axis (n), c written in place --------
__global__ void k_softmax(float* __restrict__ barr) {
    int idx = blockIdx.x * blockDim.x + threadIdx.x;   // b*512 + i
    if (idx >= BATCH * SEQ) return;
    int b = idx >> 9, i = idx & (SEQ - 1);
    float* p = barr + (size_t)b * NCAP * SEQ + i;
    float m = -1e30f;
    #pragma unroll
    for (int n = 0; n < NCAP; ++n) m = fmaxf(m, p[(size_t)n * SEQ]);
    float e[NCAP];
    float s = 0.f;
    #pragma unroll
    for (int n = 0; n < NCAP; ++n) { e[n] = __expf(p[(size_t)n * SEQ] - m); s += e[n]; }
    float inv = 1.0f / s;
    #pragma unroll
    for (int n = 0; n < NCAP; ++n) p[(size_t)n * SEQ] = e[n] * inv;
}

// ------------------------------- launcher ----------------------------------
extern "C" void kernel_launch(void* const* d_in, const int* in_sizes, int n_in,
                              void* d_out, int out_size, void* d_ws, size_t ws_size,
                              hipStream_t stream) {
    const float* u_vecs = (const float*)d_in[0];   // (64,512,512)
    const float* W      = (const float*)d_in[1];   // (1,512,2048)
    float* out = (float*)d_out;                    // (64,32,64)

    char* ws = (char*)d_ws;
    const size_t UHAT_B = (size_t)BATCH * NCAP * SEQ * DCAP * 4;  // 268 MB
    const size_t BARR_B = (size_t)BATCH * NCAP * SEQ * 4;         // 4 MB
    const size_t UV16_B = (size_t)MROW * DIN * 2;                 // 33.5 MB
    float*          uhat = (float*)ws;
    float*          barr = (float*)(ws + UHAT_B);
    unsigned short* uv16 = (unsigned short*)(ws + UHAT_B + BARR_B);
    unsigned short* wt16 = (unsigned short*)(ws + UHAT_B + BARR_B + UV16_B);

    // bf16 conversions (W also transposed to [N][K])
    {
        int n4 = (MROW * DIN) / 4;   // 4,194,304
        cvt_bf16x4<<<4096, 256, 0, stream>>>(u_vecs, uv16, n4);
        cvt_w_transpose<<<(NCOL * DIN + 255) / 256, 256, 0, stream>>>(W, wt16);
    }

    // u_hat GEMM (WMMA bf16 -> f32), transpose folded into store
    dim3 gg(MROW / BM, NCOL / BN);   // (256, 16)
    gemm_uhat<<<gg, 256, 0, stream>>>(uv16, wt16, uhat);

    const int BN_BLKS = BATCH * NCAP;            // 2048
    const int SM_BLKS = (BATCH * SEQ + 255) / 256;

    // routing iteration 0 (uniform c)
    k_out<<<BN_BLKS, 256, 0, stream>>>(uhat, nullptr, out);
    k_bupd<<<BN_BLKS, 256, 0, stream>>>(uhat, out, barr);
    k_softmax<<<SM_BLKS, 256, 0, stream>>>(barr);
    // iteration 1
    k_out<<<BN_BLKS, 256, 0, stream>>>(uhat, barr, out);
    k_bupd<<<BN_BLKS, 256, 0, stream>>>(uhat, out, barr);
    k_softmax<<<SM_BLKS, 256, 0, stream>>>(barr);
    // iteration 2 (final outputs)
    k_out<<<BN_BLKS, 256, 0, stream>>>(uhat, barr, out);
}
